// TransformerBlockQuantum_65481071407980
// MI455X (gfx1250) — matrix operand order repaired and checked by reading, not compiled, not run on hardware
//
#include <hip/hip_runtime.h>
#include <hip/hip_bf16.h>

// ---------------------------------------------------------------------------
// MI455X (gfx1250): bf16 WMMA GEMMs; tile staging via TDM (TENSOR_LOAD_TO_LDS,
// TENSORcnt) with async-to-LDS (ASYNCcnt) fallback; double-buffered K loops;
// fused epilogues. B=8 S=2048 E=1024 F=4096, M=16384.
// ---------------------------------------------------------------------------

typedef __attribute__((ext_vector_type(16))) __bf16 bf16x16;
typedef __attribute__((ext_vector_type(8)))  float  f32x8;
typedef __attribute__((ext_vector_type(4)))  unsigned int u32x4;
typedef __attribute__((ext_vector_type(8)))  int i32x8;
typedef __attribute__((ext_vector_type(4)))  int i32x4;

static constexpr int E     = 1024;
static constexpr int F     = 4096;
static constexpr int MROWS = 8 * 2048;
static constexpr int NWIRE = 32;

#define TID (threadIdx.x)

#ifndef USE_ASYNC_LDS
#define USE_ASYNC_LDS 1
#endif
#ifndef USE_TDM
#define USE_TDM 1
#endif
#if USE_TDM && __has_builtin(__builtin_amdgcn_tensor_load_to_lds)
#define HAVE_TDM 1
#else
#define HAVE_TDM 0
#endif

union FragU { uint4 u[2]; bf16x16 v; };
union Pack8 { __bf16 h[8]; uint4 u; };

__device__ __forceinline__ f32x8 fzero() {
  f32x8 z = {0.f, 0.f, 0.f, 0.f, 0.f, 0.f, 0.f, 0.f};
  return z;
}

__device__ __forceinline__ unsigned lds_off32(const void* p) {
  return (unsigned)(unsigned long long)p;  // low 32 bits = LDS offset
}

// Per-thread async fallback: 16B global -> LDS (GVS: SGPR64 base + v32 offset)
__device__ __forceinline__ void stage16(__bf16* sdst, const __bf16* gbase,
                                        int elem_off) {
#if USE_ASYNC_LDS
  asm volatile("global_load_async_to_lds_b128 %0, %1, %2"
               :
               : "v"(lds_off32(sdst)), "v"(elem_off * 2), "s"(gbase)
               : "memory");
#else
  *(uint4*)sdst = *(const uint4*)(gbase + elem_off);
#endif
}
__device__ __forceinline__ void stage_wait() {
#if USE_ASYNC_LDS
  asm volatile("s_wait_asynccnt 0x0" ::: "memory");
#endif
}

#if HAVE_TDM
// One TDM descriptor: 2D bf16 tile (tile_rows x tile_cols elems) from a row-major
// matrix (row stride = row_stride elems) into LDS at lds_off, with LDS row
// padding: pad_interval_code (2^(c+1) DWORDs per row) + pad_amount_code (c+1
// DWORDs pad). Issued per-wave; tracked by TENSORcnt.
__device__ __forceinline__ void tdm_load_2d(unsigned lds_off, const void* gsrc,
                                            unsigned tile_cols, unsigned tile_rows,
                                            unsigned row_stride,
                                            unsigned pad_interval_code,
                                            unsigned pad_amount_code) {
  unsigned long long ga = (unsigned long long)gsrc;
  u32x4 g0 = {
      1u,                                             // count=1 (user D#)
      lds_off,                                        // lds_addr
      (unsigned)(ga & 0xFFFFFFFFu),                   // global_addr[31:0]
      ((unsigned)((ga >> 32) & 0x01FFFFFFu)) | 0x80000000u  // addr[56:32], type=2
  };
  i32x8 g1 = {
      (int)((1u << 16) |                 // data_size = 1 (2 bytes)
            (1u << 20) |                 // pad_enable
            (pad_interval_code << 22) |
            (pad_amount_code << 25)),
      (int)((tile_cols & 0xFFFFu) << 16),                       // tensor_dim0 lo
      (int)(((tile_cols >> 16) & 0xFFFFu) |
            ((tile_rows & 0xFFFFu) << 16)),                     // dim0 hi, dim1 lo
      (int)(((tile_rows >> 16) & 0xFFFFu) |
            ((tile_cols & 0xFFFFu) << 16)),                     // dim1 hi, tile_dim0
      (int)(tile_rows & 0xFFFFu),                               // tile_dim1 (dim2=0)
      (int)row_stride,                                          // dim0_stride lo
      0, 0
  };
  i32x4 gz = {0, 0, 0, 0};
#if __clang_major__ >= 23
  i32x8 gz8 = {0, 0, 0, 0, 0, 0, 0, 0};
  __builtin_amdgcn_tensor_load_to_lds(g0, g1, gz, gz, gz8, 0);
#else
  __builtin_amdgcn_tensor_load_to_lds(g0, g1, gz, gz, 0);
#endif
}
#endif  // HAVE_TDM

// A fragment (16x32 bf16, MxK): lane L holds row L%16.
__device__ __forceinline__ bf16x16 ld_frag_a(const __bf16* s, int row0, int ld, int kb) {
  const int lane = TID & 31;
  const __bf16* p = s + (row0 + (lane & 15)) * ld + kb + ((lane >> 4) << 3);
  FragU f;
  f.u[0] = *(const uint4*)(p);
  f.u[1] = *(const uint4*)(p + 16);
  return f.v;
}

// B fragment (32x16 bf16, KxN): lane L holds column L%16.
__device__ __forceinline__ bf16x16 ld_frag_b(const __bf16* s, int col0, int ld, int kb) {
  const int lane = TID & 31;
  const __bf16* p = s + (col0 + (lane & 15)) * ld + kb + ((lane >> 4) << 3);
  FragU f;
  f.u[0] = *(const uint4*)(p);
  f.u[1] = *(const uint4*)(p + 16);
  return f.v;
}

__device__ __forceinline__ f32x8 wmma_bf16(bf16x16 a, bf16x16 b, f32x8 c) {
  return __builtin_amdgcn_wmma_f32_16x16x32_bf16(false, a, false, b, (short)0, c,
                                                 false, false);
}

// ---------------------------------------------------------------------------
// Vectorized f32 -> bf16 (8 elems/thread; all sizes are multiples of 8)
// ---------------------------------------------------------------------------
__global__ void __launch_bounds__(256)
k_f32_to_bf16(const float* __restrict__ in, __bf16* __restrict__ out, int n8) {
  int i = blockIdx.x * 256 + TID;
  if (i < n8) {
    float4 a = ((const float4*)in)[2 * i];
    float4 b = ((const float4*)in)[2 * i + 1];
    Pack8 p;
    p.h[0] = (__bf16)a.x; p.h[1] = (__bf16)a.y; p.h[2] = (__bf16)a.z; p.h[3] = (__bf16)a.w;
    p.h[4] = (__bf16)b.x; p.h[5] = (__bf16)b.y; p.h[6] = (__bf16)b.z; p.h[7] = (__bf16)b.w;
    ((uint4*)out)[i] = p.u;
  }
}

// ---------------------------------------------------------------------------
// Shared GEMM core pieces (128x128 C tile, K-step 32, 8 wave32s per block)
// ---------------------------------------------------------------------------
#define GEMM_PROLOG()                                                   \
  const int m0 = blockIdx.y * 128, n0 = blockIdx.x * 128;               \
  const int w = TID >> 5, lane = TID & 31;                              \
  const int wrow = (w >> 1) * 32, wcol = (w & 1) * 64;                  \
  const int wv0 = ((__builtin_amdgcn_readfirstlane((int)TID) >> 5) == 0); \
  const int ld_row = TID >> 2, ld_c = (TID & 3) << 3;                   \
  (void)wv0; (void)ld_row; (void)ld_c;

__device__ __forceinline__ void gemm_step(const __bf16* sa, const __bf16* sb,
                                          int wrow, int wcol, f32x8 acc[2][4]) {
  bf16x16 af[2], bfr[4];
#pragma unroll
  for (int i = 0; i < 2; ++i) af[i] = ld_frag_a(sa, wrow + i * 16, 40, 0);
#pragma unroll
  for (int j = 0; j < 4; ++j) bfr[j] = ld_frag_b(sb, wcol + j * 16, 40, 0);
#pragma unroll
  for (int j = 0; j < 4; ++j)
#pragma unroll
    for (int i = 0; i < 2; ++i) acc[i][j] = wmma_bf16(af[i], bfr[j], acc[i][j]);
}

// ---------------------------------------------------------------------------
// GEMM1: proj = x @ Wq^T ; qout = cos(proj) * cos(ry[n % 64])   (bf16 out)
// ---------------------------------------------------------------------------
__global__ void __launch_bounds__(256)
k_gemm_qout(const __bf16* __restrict__ xb, const __bf16* __restrict__ wq,
            const float* __restrict__ ry, __bf16* __restrict__ qout) {
  __shared__ __bf16 sA[2][128 * 40];
  __shared__ __bf16 sB[2][128 * 40];
  __shared__ float  s_cry[64];

  GEMM_PROLOG();
  if (TID < 64) s_cry[TID] = __cosf(ry[TID]);

  auto issue = [&](int k0, int buf) {
#if HAVE_TDM
    if (wv0) {
      tdm_load_2d(lds_off32(sA[buf]), xb + (size_t)m0 * E + k0, 32, 128, E, 3, 3);
      tdm_load_2d(lds_off32(sB[buf]), wq + (size_t)n0 * E + k0, 32, 128, E, 3, 3);
    }
#else
#pragma unroll
    for (int p = 0; p < 2; ++p) {
      int row = ld_row + p * 64;
      stage16(&sA[buf][row * 40 + ld_c], xb, (m0 + row) * E + k0 + ld_c);
      stage16(&sB[buf][row * 40 + ld_c], wq, (n0 + row) * E + k0 + ld_c);
    }
#endif
  };

  f32x8 acc[2][4];
#pragma unroll
  for (int i = 0; i < 2; ++i)
#pragma unroll
    for (int j = 0; j < 4; ++j) acc[i][j] = fzero();

  issue(0, 0);
  int buf = 0;
  for (int k0 = 0; k0 < E; k0 += 32) {
#if HAVE_TDM
    if (wv0) __builtin_amdgcn_s_wait_tensorcnt(0);
#else
    stage_wait();
#endif
    __syncthreads();
    if (k0 + 32 < E) issue(k0 + 32, buf ^ 1);
    gemm_step(sA[buf], sB[buf], wrow, wcol, acc);
    buf ^= 1;
  }

#pragma unroll
  for (int i = 0; i < 2; ++i)
#pragma unroll
    for (int j = 0; j < 4; ++j) {
      int mb = m0 + wrow + i * 16 + ((lane >> 4) << 3);
      int n  = n0 + wcol + j * 16 + (lane & 15);
      float cr = s_cry[n & 63];
#pragma unroll
      for (int v = 0; v < 8; ++v)
        qout[(size_t)(mb + v) * E + n] = (__bf16)(__cosf(acc[i][j][v]) * cr);
    }
}

// ---------------------------------------------------------------------------
// GEMM2: res = qout @ Wc^T + combine_b + x   (f32 out, pre-LN1 residual)
// ---------------------------------------------------------------------------
__global__ void __launch_bounds__(256)
k_gemm_attn(const __bf16* __restrict__ qo, const __bf16* __restrict__ wc,
            const float* __restrict__ cb, const float* __restrict__ x,
            float* __restrict__ res) {
  __shared__ __bf16 sA[2][128 * 40];
  __shared__ __bf16 sB[2][128 * 40];

  GEMM_PROLOG();

  auto issue = [&](int k0, int buf) {
#if HAVE_TDM
    if (wv0) {
      tdm_load_2d(lds_off32(sA[buf]), qo + (size_t)m0 * E + k0, 32, 128, E, 3, 3);
      tdm_load_2d(lds_off32(sB[buf]), wc + (size_t)n0 * E + k0, 32, 128, E, 3, 3);
    }
#else
#pragma unroll
    for (int p = 0; p < 2; ++p) {
      int row = ld_row + p * 64;
      stage16(&sA[buf][row * 40 + ld_c], qo, (m0 + row) * E + k0 + ld_c);
      stage16(&sB[buf][row * 40 + ld_c], wc, (n0 + row) * E + k0 + ld_c);
    }
#endif
  };

  f32x8 acc[2][4];
#pragma unroll
  for (int i = 0; i < 2; ++i)
#pragma unroll
    for (int j = 0; j < 4; ++j) acc[i][j] = fzero();

  issue(0, 0);
  int buf = 0;
  for (int k0 = 0; k0 < E; k0 += 32) {
#if HAVE_TDM
    if (wv0) __builtin_amdgcn_s_wait_tensorcnt(0);
#else
    stage_wait();
#endif
    __syncthreads();
    if (k0 + 32 < E) issue(k0 + 32, buf ^ 1);
    gemm_step(sA[buf], sB[buf], wrow, wcol, acc);
    buf ^= 1;
  }

#pragma unroll
  for (int i = 0; i < 2; ++i)
#pragma unroll
    for (int j = 0; j < 4; ++j) {
      int mb = m0 + wrow + i * 16 + ((lane >> 4) << 3);
      int n  = n0 + wcol + j * 16 + (lane & 15);
      float bias = cb[n];
#pragma unroll
      for (int v = 0; v < 8; ++v) {
        size_t o = (size_t)(mb + v) * E + n;
        res[o] = acc[i][j][v] + bias + x[o];
      }
    }
}

// ---------------------------------------------------------------------------
// LayerNorm over E=1024, one row per 256-thread block.
// ---------------------------------------------------------------------------
__global__ void __launch_bounds__(256)
k_layernorm(const float* __restrict__ in, const float* __restrict__ g,
            const float* __restrict__ b, float* __restrict__ out) {
  __shared__ float rs[256], rq[256];
  size_t row = blockIdx.x;
  const float4 v = ((const float4*)(in + row * E))[TID];
  rs[TID] = v.x + v.y + v.z + v.w;
  rq[TID] = v.x * v.x + v.y * v.y + v.z * v.z + v.w * v.w;
  __syncthreads();
  for (int o = 128; o > 0; o >>= 1) {
    if (TID < o) { rs[TID] += rs[TID + o]; rq[TID] += rq[TID + o]; }
    __syncthreads();
  }
  const float mean = rs[0] * (1.f / E);
  const float var  = rq[0] * (1.f / E) - mean * mean;
  const float inv  = rsqrtf(var + 1e-5f);
  const float4 gg = ((const float4*)g)[TID];
  const float4 bb = ((const float4*)b)[TID];
  float4 o4;
  o4.x = (v.x - mean) * inv * gg.x + bb.x;
  o4.y = (v.y - mean) * inv * gg.y + bb.y;
  o4.z = (v.z - mean) * inv * gg.z + bb.z;
  o4.w = (v.w - mean) * inv * gg.w + bb.w;
  ((float4*)(out + row * E))[TID] = o4;
}

// ---------------------------------------------------------------------------
// Fused FFN: qffn = cos(h[:, :32]); hid = relu(qffn @ lin1^T + b1) (LDS only);
//            res = hid @ lin2^T + b2 + h.  128x128 out tile, F in 128 chunks.
// ---------------------------------------------------------------------------
__global__ void __launch_bounds__(256)
k_ffn(const float* __restrict__ h, const __bf16* __restrict__ l1,
      const __bf16* __restrict__ l2, const float* __restrict__ b1,
      const float* __restrict__ b2, float* __restrict__ res) {
  extern __shared__ char smem_raw[];
  __bf16* sQ  = (__bf16*)smem_raw;   // 128 x 40  (qffn, K=32 + pad)
  __bf16* sB1 = sQ  + 128 * 40;      // 128 x 40  (lin1 tile)
  __bf16* sH  = sB1 + 128 * 40;      // 128 x 136 (hid chunk, K=128 + pad)
  __bf16* sB2 = sH  + 128 * 136;     // 128 x 136 (lin2 tile)

  const int e0 = blockIdx.x * 128, m0 = blockIdx.y * 128;
  const int w = TID >> 5, lane = TID & 31;
  const int wrow = (w >> 1) * 32, wcol = (w & 1) * 64;
  const int wv0 = ((__builtin_amdgcn_readfirstlane((int)TID) >> 5) == 0);
  (void)wv0;

  // Stage 1: qffn = cos(h[:, :32]) as bf16 A-tile
  for (int idx = TID; idx < 128 * 8; idx += 256) {
    int row = idx >> 3, qd = (idx & 7) << 2;
    float4 hv = *(const float4*)(h + (size_t)(m0 + row) * E + qd);
    __bf16* d = &sQ[row * 40 + qd];
    d[0] = (__bf16)__cosf(hv.x);
    d[1] = (__bf16)__cosf(hv.y);
    d[2] = (__bf16)__cosf(hv.z);
    d[3] = (__bf16)__cosf(hv.w);
  }

  f32x8 acc[2][4];
#pragma unroll
  for (int i = 0; i < 2; ++i)
#pragma unroll
    for (int j = 0; j < 4; ++j) acc[i][j] = fzero();

  for (int fc = 0; fc < F; fc += 128) {
    __syncthreads();  // prev GEMM4 done before tile overwrite
#if HAVE_TDM
    if (wv0) {
      tdm_load_2d(lds_off32(sB1), l1 + (size_t)fc * NWIRE, 32, 128, NWIRE, 3, 3);
      tdm_load_2d(lds_off32(sB2), l2 + (size_t)e0 * F + fc, 128, 128, F, 5, 3);
      __builtin_amdgcn_s_wait_tensorcnt(0);
    }
#else
#pragma unroll
    for (int p = 0; p < 2; ++p) {
      int idx = TID + p * 256;
      int rowf = idx >> 2, c = (idx & 3) << 3;
      stage16(&sB1[rowf * 40 + c], l1, (fc + rowf) * NWIRE + c);
    }
#pragma unroll
    for (int p = 0; p < 8; ++p) {
      int idx = TID + p * 256;
      int rowe = idx >> 4, c = (idx & 15) << 3;
      stage16(&sB2[rowe * 136 + c], l2, (e0 + rowe) * F + fc + c);
    }
    stage_wait();
#endif
    __syncthreads();

    // GEMM3 (K=32): hid chunk -> +bias, relu -> bf16 into sH
#pragma unroll
    for (int i = 0; i < 2; ++i) {
      bf16x16 aq = ld_frag_a(sQ, wrow + i * 16, 40, 0);
#pragma unroll
      for (int j = 0; j < 4; ++j) {
        bf16x16 bq = ld_frag_b(sB1, wcol + j * 16, 40, 0);
        f32x8 hh = wmma_bf16(aq, bq, fzero());
        int ml = wrow + i * 16 + ((lane >> 4) << 3);
        int kl = wcol + j * 16 + (lane & 15);
        float bias = b1[fc + kl];
#pragma unroll
        for (int v = 0; v < 8; ++v) {
          float val = hh[v] + bias;
          sH[(ml + v) * 136 + kl] = (__bf16)(val > 0.f ? val : 0.f);
        }
      }
    }
    __syncthreads();

    // GEMM4 accumulate: acc += hid_chunk @ lin2_tile^T  (4 K-steps of 32)
#pragma unroll
    for (int kk = 0; kk < 4; ++kk) {
      bf16x16 af[2], bfr[4];
#pragma unroll
      for (int i = 0; i < 2; ++i) af[i] = ld_frag_a(sH, wrow + i * 16, 136, kk * 32);
#pragma unroll
      for (int j = 0; j < 4; ++j) bfr[j] = ld_frag_b(sB2, wcol + j * 16, 136, kk * 32);
#pragma unroll
      for (int j = 0; j < 4; ++j)
#pragma unroll
        for (int i = 0; i < 2; ++i) acc[i][j] = wmma_bf16(af[i], bfr[j], acc[i][j]);
    }
  }

  // Epilogue: + lin2_b + h residual -> f32
#pragma unroll
  for (int i = 0; i < 2; ++i)
#pragma unroll
    for (int j = 0; j < 4; ++j) {
      int mb = m0 + wrow + i * 16 + ((lane >> 4) << 3);
      int n  = e0 + wcol + j * 16 + (lane & 15);
      float bias = b2[n];
#pragma unroll
      for (int v = 0; v < 8; ++v) {
        size_t o = (size_t)(mb + v) * E + n;
        res[o] = acc[i][j][v] + bias + h[o];
      }
    }
}

// ---------------------------------------------------------------------------
// Launch
// ---------------------------------------------------------------------------
extern "C" void kernel_launch(void* const* d_in, const int* in_sizes, int n_in,
                              void* d_out, int out_size, void* d_ws, size_t ws_size,
                              hipStream_t stream) {
  const float* x     = (const float*)d_in[0];
  const float* wq_f  = (const float*)d_in[1];
  const float* ry    = (const float*)d_in[2];
  const float* wc_f  = (const float*)d_in[3];
  const float* cb    = (const float*)d_in[4];
  const float* ln1g  = (const float*)d_in[5];
  const float* ln1b  = (const float*)d_in[6];
  const float* l1_f  = (const float*)d_in[7];
  const float* l1b   = (const float*)d_in[8];
  const float* l2_f  = (const float*)d_in[9];
  const float* l2b   = (const float*)d_in[10];
  const float* ln2g  = (const float*)d_in[11];
  const float* ln2b  = (const float*)d_in[12];
  float* out = (float*)d_out;

  size_t off = 0;
  auto carve = [&](size_t bytes) {
    void* p = (char*)d_ws + off;
    off += (bytes + 255) & ~(size_t)255;
    return p;
  };
  __bf16* xb   = (__bf16*)carve((size_t)MROWS * E * 2);
  __bf16* wqb  = (__bf16*)carve((size_t)E * E * 2);
  __bf16* wcb  = (__bf16*)carve((size_t)E * E * 2);
  __bf16* l1bw = (__bf16*)carve((size_t)F * NWIRE * 2);
  __bf16* l2bw = (__bf16*)carve((size_t)E * F * 2);
  __bf16* qout = (__bf16*)carve((size_t)MROWS * E * 2);
  float*  resb = (float*)carve((size_t)MROWS * E * 4);
  float*  hbuf = (float*)carve((size_t)MROWS * E * 4);

  auto cvt = [&](const float* src, __bf16* dst, size_t n) {
    int n8 = (int)(n / 8);
    k_f32_to_bf16<<<(n8 + 255) / 256, 256, 0, stream>>>(src, dst, n8);
  };
  cvt(x,    xb,   (size_t)MROWS * E);
  cvt(wq_f, wqb,  (size_t)E * E);
  cvt(wc_f, wcb,  (size_t)E * E);
  cvt(l1_f, l1bw, (size_t)F * NWIRE);
  cvt(l2_f, l2bw, (size_t)E * F);

  dim3 gemm_grid(E / 128, MROWS / 128);

  k_gemm_qout<<<gemm_grid, 256, 0, stream>>>(xb, wqb, ry, qout);
  k_gemm_attn<<<gemm_grid, 256, 0, stream>>>(qout, wcb, cb, x, resb);
  k_layernorm<<<MROWS, 256, 0, stream>>>(resb, ln1g, ln1b, hbuf);

  const size_t smem = (size_t)(128 * 40 + 128 * 40 + 128 * 136 + 128 * 136) * 2;
  (void)hipFuncSetAttribute((const void*)k_ffn,
                            hipFuncAttributeMaxDynamicSharedMemorySize,
                            (int)smem);
  k_ffn<<<gemm_grid, 256, smem, stream>>>(hbuf, l1bw, l2bw, l1b, l2b, resb);

  k_layernorm<<<MROWS, 256, 0, stream>>>(resb, ln2g, ln2b, out);
}
